// PSMMix3dEmbedding_40596030881943
// MI455X (gfx1250) — compile-verified, matrix operand order checked by reading, not run on hardware
//
#include <hip/hip_runtime.h>
#include <hip/hip_bf16.h>
#include <math.h>

// ---------------------------------------------------------------------------
// Problem constants (match reference)
// ---------------------------------------------------------------------------
#define B_  2
#define L_  512
#define K_  128
#define H_  32
#define D_  1024

#define SQRT_2PI_   2.5066282746310002f
#define MIN_F32_    (-3.4028234663852886e38f)
#define SCALING_    0.08838834764831845f   // 128^-0.5
#define INV_SQRT2_  0.70710678118654752f
#define LOG2E_      1.4426950408889634f
#define SQ_HL2E_    0.8493220567168383f    // sqrt(0.5*log2(e))

typedef __bf16 bf16_t;
typedef __attribute__((ext_vector_type(16))) bf16_t v16bf;
typedef __attribute__((ext_vector_type(8)))  float  v8f;

union AFrag { v16bf v; bf16_t e[16]; };
union CFrag { v8f   v; float  e[8];  };

// single v_exp_f32 (hardware exp2)
__device__ __forceinline__ float fast_exp2(float x) {
    return __builtin_amdgcn_exp2f(x);
}

// ---------------------------------------------------------------------------
// Prep kernels: weight conversion to bf16, pos_emb precompute, ef zeroing
// ---------------------------------------------------------------------------
__global__ void k_prep_w(const float* __restrict__ w1, const float* __restrict__ w2,
                         bf16_t* __restrict__ w1b, bf16_t* __restrict__ w2b) {
    int i = blockIdx.x * 256 + threadIdx.x;          // 64 blocks -> 16384 threads
    if (i < K_ * K_) w1b[i] = (bf16_t)w1[i];
    if (i < H_ * K_) w2b[i] = (bf16_t)w2[i];
}

// Wc[d][k] = k<128 ? pos_feature_emb_w[d][k] : pos_proj_w[d][k-128]   (1024 x 256)
__global__ void k_prep_wc(const float* __restrict__ pfw, const float* __restrict__ ppw,
                          bf16_t* __restrict__ wc) {
    int i = blockIdx.x * 256 + threadIdx.x;          // 1024 blocks -> 262144
    int d = i >> 8, k = i & 255;
    float v = (k < K_) ? pfw[d * K_ + k] : ppw[d * K_ + (k - K_)];
    wc[i] = (bf16_t)v;
}

// pos_emb[b,m,k] (masked) and ef_ws zero-init (both B*L*K = 131072 elements)
__global__ void k_prep_pe(const float* __restrict__ pos, const float* __restrict__ pew,
                          const unsigned char* __restrict__ padm,
                          float* __restrict__ pos_emb, float* __restrict__ ef_ws) {
    int i = blockIdx.x * 256 + threadIdx.x;          // 512 blocks
    int row = i >> 7, k = i & 127;                   // row = b*L + m
    ef_ws[i] = 0.0f;
    float v = 0.0f;
    if (!padm[row]) {
        v = pos[row * 3 + 0] * pew[k * 3 + 0]
          + pos[row * 3 + 1] * pew[k * 3 + 1]
          + pos[row * 3 + 2] * pew[k * 3 + 2];
    }
    pos_emb[i] = v;
}

// ---------------------------------------------------------------------------
// Main fused kernel: gaussian edge features -> GEMM1(W1)+GELU -> GEMM2(W2),
// masking, graph_attn_bias output, head-sum dist, masked ef reduction.
// Block = 256 threads = 8 waves; each wave = 16 edge rows; block = 128
// consecutive m for one (b,l).  Grid = B*L*L/128 = 4096.
// ---------------------------------------------------------------------------
__global__ __launch_bounds__(256, 1)
void k_edge_main(const float* __restrict__ pos, const int* __restrict__ nte,
                 const unsigned char* __restrict__ adj,
                 const unsigned char* __restrict__ padm,
                 const unsigned char* __restrict__ molm,
                 const unsigned char* __restrict__ clnm,
                 const float* __restrict__ gbf_means, const float* __restrict__ gbf_stds,
                 const float* __restrict__ gbf_mul,   const float* __restrict__ gbf_bias,
                 const bf16_t* __restrict__ w1b, const float* __restrict__ b1,
                 const bf16_t* __restrict__ w2b, const float* __restrict__ b2,
                 float* __restrict__ gab_out,   // [B,H,L,L]
                 float* __restrict__ dist_ws,   // [B,L,L]
                 float* __restrict__ ef_ws)     // [B,L,K]
{
    extern __shared__ char smem[];
    bf16_t* w1s = (bf16_t*)(smem);               // 128x128 bf16 = 32768 B
    bf16_t* w2s = (bf16_t*)(smem + 32768);       //  32x128 bf16 =  8192 B
    bf16_t* hs  = (bf16_t*)(smem + 40960);       // 8 waves * 16x128 bf16 = 32768 B
    float*  gabs = (float*)(smem + 73728);       // 32 n x 128 m f32 = 16384 B
    float*  efs  = (float*)(smem + 90112);       // 128 f32
    float*  gA   = (float*)(smem + 90624);       // sqrt(0.5*log2e)/std    128
    float*  gB   = (float*)(smem + 91136);       // -mean * gA             128
    float*  gcs  = (float*)(smem + 91648);       // 1/(sqrt(2pi)*std)      128
    unsigned int* flg = (unsigned int*)(smem + 92160); // 128 per-m flags

    const int t    = threadIdx.x;
    const int wave = t >> 5;
    const int lane = t & 31;
    const int rt   = lane & 15;
    const int half = lane >> 4;

    // --- stage bf16 weights + gaussian params into LDS -----------------------
    for (int i = t; i < (K_ * K_) / 2; i += 256)
        ((unsigned int*)w1s)[i] = ((const unsigned int*)w1b)[i];
    for (int i = t; i < (H_ * K_) / 2; i += 256)
        ((unsigned int*)w2s)[i] = ((const unsigned int*)w2b)[i];
    if (t < K_) {
        const float s = fabsf(gbf_stds[t]) + 1e-5f;
        const float a = SQ_HL2E_ / s;
        gA[t]  = a;
        gB[t]  = -gbf_means[t] * a;
        gcs[t] = 1.0f / (SQRT_2PI_ * s);
        efs[t] = 0.0f;
    }

    // --- decode tile coordinates --------------------------------------------
    const int blk = blockIdx.x;
    const int b   = blk / ((L_ * L_) / 128);
    const int rem = blk - b * ((L_ * L_) / 128);
    const int l   = rem / (L_ / 128);
    const int m0  = (rem % (L_ / 128)) * 128;
    const int m   = m0 + wave * 16 + rt;           // this lane's edge row (m)
    const bool padL = padm[b * L_ + l] != 0;

    // --- per-lane edge scalars ----------------------------------------------
    const float pl0 = pos[(b * L_ + l) * 3 + 0];
    const float pl1 = pos[(b * L_ + l) * 3 + 1];
    const float pl2 = pos[(b * L_ + l) * 3 + 2];
    const float pm0 = pos[(b * L_ + m) * 3 + 0];
    const float pm1 = pos[(b * L_ + m) * 3 + 1];
    const float pm2 = pos[(b * L_ + m) * 3 + 2];
    const float dx = pl0 - pm0, dy = pl1 - pm1, dz = pl2 - pm2;
    const float dist0 = sqrtf(dx * dx + dy * dy + dz * dz + 1e-12f);

    const long eidx = ((long)(b * L_ + l) * L_ + m) * 2;
    const int e0 = nte[eidx], e1 = nte[eidx + 1];
    const float mul = gbf_mul[e0] + gbf_mul[e1];
    const float bia = gbf_bias[e0] + gbf_bias[e1];
    const float x = mul * dist0 + bia;

    const bool padM = padm[b * L_ + m] != 0;
    const bool adjF = (adj[(long)(b * L_ + l) * L_ + m] != 0)
                   || (molm[b * L_ + l] == 0) || (clnm[b * L_ + l] != 0);
    if (half == 0) flg[wave * 16 + rt] = (padM ? 1u : 0u) | (adjF ? 2u : 0u);
    __syncthreads();

    // --- GEMM1: edge_feature(16x128) x W1^T -> h(16x128), f32 accum ----------
    CFrag acc1[8];
#pragma unroll
    for (int nt = 0; nt < 8; ++nt)
#pragma unroll
        for (int i = 0; i < 8; ++i) acc1[nt].e[i] = 0.0f;

#pragma unroll
    for (int kb = 0; kb < 4; ++kb) {
        const int k0 = kb * 32 + half * 16;
        AFrag a;
        float fv[16];
#pragma unroll
        for (int e = 0; e < 16; ++e) {
            const int k = k0 + e;
            // exp(-0.5*((x-mu)/sig)^2)/(sqrt(2pi)sig) = exp2(-(x*A+B)^2)*C
            const float u = fmaf(x, gA[k], gB[k]);
            const float g = fast_exp2(-(u * u)) * gcs[k];
            fv[e]  = g;
            a.e[e] = (bf16_t)g;
        }
        if (!padM) {                      // masked ef reduction into LDS
#pragma unroll
            for (int e = 0; e < 16; ++e) atomicAdd(&efs[k0 + e], fv[e]);
        }
#pragma unroll
        for (int nt = 0; nt < 8; ++nt) {
            AFrag bf;
            bf.v = *(const v16bf*)&w1s[(nt * 16 + rt) * K_ + k0];
            acc1[nt].v = __builtin_amdgcn_wmma_f32_16x16x32_bf16(
                false, a.v, false, bf.v, (short)0, acc1[nt].v, false, false);
        }
    }

    // --- bias + exact GELU, restage h as bf16 A-fragments --------------------
#pragma unroll
    for (int nt = 0; nt < 8; ++nt) {
        const float bb = b1[nt * 16 + rt];
#pragma unroll
        for (int i = 0; i < 8; ++i) {
            const float v = acc1[nt].e[i] + bb;
            const float g = 0.5f * v * (1.0f + erff(v * INV_SQRT2_));
            hs[(wave * 16 + i + 8 * half) * K_ + nt * 16 + rt] = (bf16_t)g;
        }
    }
    __syncthreads();

    // --- GEMM2: h(16x128) x W2^T -> gab(16x32) -------------------------------
    CFrag acc2[2];
#pragma unroll
    for (int nt = 0; nt < 2; ++nt)
#pragma unroll
        for (int i = 0; i < 8; ++i) acc2[nt].e[i] = 0.0f;

#pragma unroll
    for (int kb = 0; kb < 4; ++kb) {
        const int k0 = kb * 32 + half * 16;
        AFrag a2;
        a2.v = *(const v16bf*)&hs[(wave * 16 + rt) * K_ + k0];
#pragma unroll
        for (int nt = 0; nt < 2; ++nt) {
            AFrag bf;
            bf.v = *(const v16bf*)&w2s[(nt * 16 + rt) * K_ + k0];
            acc2[nt].v = __builtin_amdgcn_wmma_f32_16x16x32_bf16(
                false, a2.v, false, bf.v, (short)0, acc2[nt].v, false, false);
        }
    }

    // --- masking -> LDS transpose buffer (n-major, m-minor) ------------------
#pragma unroll
    for (int nt = 0; nt < 2; ++nt) {
        const float bb = b2[nt * 16 + rt];
#pragma unroll
        for (int i = 0; i < 8; ++i) {
            const int M = i + 8 * half;
            const unsigned f = flg[wave * 16 + M];
            const float v = acc2[nt].e[i] + bb;
            float outv;
            if (padL)                         outv = 0.0f;
            else if ((f & 1u) || !(f & 2u))   outv = -INFINITY;
            else                              outv = v;
            gabs[(nt * 16 + rt) * 128 + wave * 16 + M] = outv;
        }
    }
    __syncthreads();

    // --- coalesced graph_attn_bias writes: [B,H,L,L] -------------------------
    const long obase = ((long)b * H_) * (long)L_ * L_ + (long)l * L_ + m0;
#pragma unroll
    for (int s = 0; s < 16; ++s) {
        const int idx = t + 256 * s;         // 0..4095
        const int n = idx >> 7, mm = idx & 127;
        gab_out[obase + (long)n * L_ * L_ + mm] = gabs[n * 128 + mm];
    }

    // --- dist (head sum + masks) and ef flush --------------------------------
    if (t < 128) {
        const unsigned f = flg[t];
        float dv;
        if (padL || (f & 1u) || !(f & 2u)) {
            dv = MIN_F32_;
        } else {
            float s = 0.0f;
#pragma unroll
            for (int n = 0; n < H_; ++n) s += gabs[n * 128 + t];
            dv = s;
        }
        dist_ws[(long)(b * L_ + l) * L_ + m0 + t] = dv;
        atomicAdd(&ef_ws[(b * L_ + l) * K_ + t], efs[t]);
    }
}

// ---------------------------------------------------------------------------
// Softmax over m (with max-subtract, matching jax) + weighted pos_emb agg.
// One block (128 threads) per (b,l) row.
// ---------------------------------------------------------------------------
__global__ __launch_bounds__(128, 2)
void k_softmax_agg(const float* __restrict__ dist_ws,
                   const float* __restrict__ pos_emb,
                   float* __restrict__ pfek) {
    __shared__ float red[128];
    __shared__ float wls[L_];
    const int t = threadIdx.x;
    const int row = blockIdx.x;                 // b*L + l
    const int b = row / L_;
    const float* dr = dist_ws + (long)row * L_;

    float v[4];
    float mx = -INFINITY;
#pragma unroll
    for (int j = 0; j < 4; ++j) { v[j] = dr[t + 128 * j]; mx = fmaxf(mx, v[j]); }
    red[t] = mx; __syncthreads();
    for (int s = 64; s > 0; s >>= 1) { if (t < s) red[t] = fmaxf(red[t], red[t + s]); __syncthreads(); }
    mx = red[0]; __syncthreads();

    // exp((v-mx)*S) as a single hardware exp2 with folded scale
    const float S2 = SCALING_ * LOG2E_;
    float sum = 0.0f;
#pragma unroll
    for (int j = 0; j < 4; ++j) {
        const float e = fast_exp2((v[j] - mx) * S2);
        wls[t + 128 * j] = e;
        sum += e;
    }
    red[t] = sum; __syncthreads();
    for (int s = 64; s > 0; s >>= 1) { if (t < s) red[t] += red[t + s]; __syncthreads(); }
    const float inv = 1.0f / red[0];

    // pfe_k[k=t] = (1/Z) * sum_m w[m] * pos_emb[b,m,k]
    float acc = 0.0f;
    const float* pe = pos_emb + (long)b * L_ * K_ + t;
    for (int m2 = 0; m2 < L_; ++m2) acc += wls[m2] * pe[(long)m2 * K_];
    pfek[(long)row * K_ + t] = acc * inv;
}

// ---------------------------------------------------------------------------
// Final GEMM: [pfe_k | ef] (1024 x 256, f32->bf16) x Wc^T (1024 x 256 bf16)
// -> pfe [1024 x 1024] + bias, pad rows zeroed.  8 waves/block, each a 16x16
// tile; grid = 64 m-tiles * 8 col-groups.
// ---------------------------------------------------------------------------
__global__ __launch_bounds__(256, 1)
void k_final(const float* __restrict__ pfek, const float* __restrict__ ef_ws,
             const bf16_t* __restrict__ wc, const float* __restrict__ ppb,
             const unsigned char* __restrict__ padm, float* __restrict__ outp) {
    const int t = threadIdx.x, wave = t >> 5, lane = t & 31;
    const int rt = lane & 15, half = lane >> 4;
    const int bx = blockIdx.x;
    const int mt = bx >> 3, cg = bx & 7;
    const int rowbase = mt * 16;
    const int dbase = cg * 128 + wave * 16;

    CFrag acc;
#pragma unroll
    for (int i = 0; i < 8; ++i) acc.e[i] = 0.0f;

#pragma unroll
    for (int kb = 0; kb < 8; ++kb) {
        const int k0 = kb * 32 + half * 16;
        const float* src = (k0 < K_)
            ? (pfek  + (long)(rowbase + rt) * K_ + k0)
            : (ef_ws + (long)(rowbase + rt) * K_ + (k0 - K_));
        AFrag a;
#pragma unroll
        for (int e = 0; e < 16; ++e) a.e[e] = (bf16_t)src[e];
        AFrag bf;
        bf.v = *(const v16bf*)&wc[(long)(dbase + rt) * 256 + k0];
        acc.v = __builtin_amdgcn_wmma_f32_16x16x32_bf16(
            false, a.v, false, bf.v, (short)0, acc.v, false, false);
    }

    const float bb = ppb[dbase + rt];
#pragma unroll
    for (int i = 0; i < 8; ++i) {
        const int M = i + 8 * half;
        const int row = rowbase + M;                  // = b*L + l
        float v = acc.e[i] + bb;
        if (padm[row]) v = 0.0f;
        outp[(long)row * D_ + dbase + rt] = v;
    }
}

// ---------------------------------------------------------------------------
// Launch
// ---------------------------------------------------------------------------
extern "C" void kernel_launch(void* const* d_in, const int* in_sizes, int n_in,
                              void* d_out, int out_size, void* d_ws, size_t ws_size,
                              hipStream_t stream) {
    const float*         pos  = (const float*)d_in[0];
    const int*           nte  = (const int*)d_in[1];
    const unsigned char* adj  = (const unsigned char*)d_in[2];
    const unsigned char* padm = (const unsigned char*)d_in[3];
    const unsigned char* molm = (const unsigned char*)d_in[4];
    const unsigned char* clnm = (const unsigned char*)d_in[5];
    const float*         pew  = (const float*)d_in[6];   // pos_emb_w [K,3]
    const float*         pfw  = (const float*)d_in[7];   // pos_feature_emb_w [D,K]
    const float*         gme  = (const float*)d_in[8];
    const float*         gst  = (const float*)d_in[9];
    const float*         gmu  = (const float*)d_in[10];
    const float*         gbi  = (const float*)d_in[11];
    const float*         w1   = (const float*)d_in[12];
    const float*         b1   = (const float*)d_in[13];
    const float*         w2   = (const float*)d_in[14];
    const float*         b2   = (const float*)d_in[15];
    const float*         ppw  = (const float*)d_in[16];  // pos_proj_w [D,K]
    const float*         ppb  = (const float*)d_in[17];

    char* ws = (char*)d_ws;
    bf16_t* w1b     = (bf16_t*)(ws + 0);          // 32768 B
    bf16_t* w2b     = (bf16_t*)(ws + 32768);      //  8192 B
    bf16_t* wc      = (bf16_t*)(ws + 40960);      // 524288 B
    float*  pos_emb = (float*) (ws + 565248);     // 524288 B
    float*  dist_ws = (float*) (ws + 1089536);    // 2097152 B
    float*  ef_ws   = (float*) (ws + 3186688);    // 524288 B
    float*  pfek    = (float*) (ws + 3710976);    // 524288 B  (total ~4.04 MB)

    float* pfe_out = (float*)d_out;                       // B*L*D
    float* gab_out = pfe_out + (size_t)B_ * L_ * D_;      // B*H*L*L

    k_prep_w <<<64,   256, 0, stream>>>(w1, w2, w1b, w2b);
    k_prep_wc<<<1024, 256, 0, stream>>>(pfw, ppw, wc);
    k_prep_pe<<<512,  256, 0, stream>>>(pos, pew, padm, pos_emb, ef_ws);

    const size_t smem = 92672;  // 90.5 KB dynamic LDS (within 320 KB/WG)
    k_edge_main<<<(B_ * L_ * L_) / 128, 256, smem, stream>>>(
        pos, nte, adj, padm, molm, clnm, gme, gst, gmu, gbi,
        w1b, b1, w2b, b2, gab_out, dist_ws, ef_ws);

    k_softmax_agg<<<B_ * L_, 128, 0, stream>>>(dist_ws, pos_emb, pfek);

    k_final<<<(B_ * L_ / 16) * (D_ / 128), 256, 0, stream>>>(
        pfek, ef_ws, wc, ppb, padm, pfe_out);
}
